// Zencoder_48808008352091
// MI455X (gfx1250) — matrix operand order repaired and checked by reading, not compiled, not run on hardware
//
#include <hip/hip_runtime.h>
#include <hip/hip_bf16.h>

typedef _Float16 half_t;
typedef __attribute__((ext_vector_type(16))) _Float16 v16h;
typedef __attribute__((ext_vector_type(8)))  _Float16 v8h;
typedef __attribute__((ext_vector_type(8)))  float    v8f;
typedef int v4i __attribute__((ext_vector_type(4)));
typedef __attribute__((address_space(1))) v4i* g_v4i_p;   // global (AS1) int4*
typedef __attribute__((address_space(3))) v4i* l_v4i_p;   // LDS    (AS3) int4*

union V16U { v16h v; v8h h[2]; };

// ---- gfx1250 async global->LDS copy (ASYNCcnt path), compile-safe guarded ----
#if __has_builtin(__builtin_amdgcn_global_load_async_to_lds_b128) && \
    __has_builtin(__builtin_amdgcn_s_wait_asynccnt)
#define CP16(dst, src) __builtin_amdgcn_global_load_async_to_lds_b128(          \
    (g_v4i_p)(src), (l_v4i_p)(dst), 0, 0)
#define ASYNC_WAIT_KEEP(n) __builtin_amdgcn_s_wait_asynccnt(n)
#define ASYNC_WAIT_ALL()   __builtin_amdgcn_s_wait_asynccnt(0)
#else
#define CP16(dst, src) (*(v8h*)(dst) = *(const v8h*)(src))
#define ASYNC_WAIT_KEEP(n)
#define ASYNC_WAIT_ALL()
#endif

__device__ __forceinline__ int reflect_idx(int i, int n) {
    return i < 0 ? -i : (i >= n ? 2 * n - 2 - i : i);
}

__device__ __forceinline__ float fast_tanh(float v) {
    v = fminf(fmaxf(v, -15.f), 15.f);
    float e = __expf(2.f * v);
    return __fdividef(e - 1.f, e + 1.f);
}

// ---------------- conv1: 3->32, k3, reflect pad 1, 256x256 (bandwidth-bound) ----------------
__global__ void conv1_k(const float* __restrict__ x, const float* __restrict__ w,
                        const float* __restrict__ bias, float* __restrict__ y) {
    int idx = blockIdx.x * 256 + threadIdx.x;            // B*32*256*256 = 2^24
    if (idx >= 8 * 32 * 256 * 256) return;
    int xw = idx & 255, yh = (idx >> 8) & 255, co = (idx >> 16) & 31, b = idx >> 21;
    float acc = bias[co];
    for (int ci = 0; ci < 3; ++ci)
        for (int ky = 0; ky < 3; ++ky) {
            int ry = reflect_idx(yh + ky - 1, 256);
            for (int kx = 0; kx < 3; ++kx) {
                int rx = reflect_idx(xw + kx - 1, 256);
                acc += x[((b * 3 + ci) * 256 + ry) * 256 + rx] *
                       w[((co * 3 + ci) * 3 + ky) * 3 + kx];
            }
        }
    y[idx] = acc;
}

// ---------------- instance norm: stats ----------------
__global__ void inorm_stats_k(const float* __restrict__ x, float* __restrict__ mean,
                              float* __restrict__ rstd, int HW) {
    int bc = blockIdx.x;
    const float* p = x + (long)bc * HW;
    float s = 0.f, s2 = 0.f;
    for (int i = threadIdx.x; i < HW; i += 256) { float v = p[i]; s += v; s2 += v * v; }
    __shared__ float sh[256], sh2[256];
    sh[threadIdx.x] = s; sh2[threadIdx.x] = s2; __syncthreads();
    for (int o = 128; o > 0; o >>= 1) {
        if (threadIdx.x < o) { sh[threadIdx.x] += sh[threadIdx.x + o]; sh2[threadIdx.x] += sh2[threadIdx.x + o]; }
        __syncthreads();
    }
    if (threadIdx.x == 0) {
        float m = sh[0] / HW;
        float v = sh2[0] / HW - m * m;
        mean[bc] = m;
        rstd[bc] = rsqrtf(fmaxf(v, 0.f) + 1e-5f);
    }
}

// IN+lrelu -> zero-padded f16 NHWC (pad 1 on each side). Generic over C, H.
__global__ void inorm_lrelu_pad_k(const float* __restrict__ y, const float* __restrict__ mean,
                                  const float* __restrict__ rstd, half_t* __restrict__ xp,
                                  int C, int H, long total) {
    long idx = (long)blockIdx.x * 256 + threadIdx.x;     // b,c,yh,xw over NCHW
    if (idx >= total) return;
    int xw = idx % H; long t = idx / H;
    int yh = t % H; t /= H;
    int c = t % C; int b = t / C;
    int bc = b * C + c;
    int HP = H + 2;
    float v = (y[idx] - mean[bc]) * rstd[bc];
    v = v > 0.f ? v : 0.2f * v;
    xp[(((long)(b * HP + yh + 1) * HP) + xw + 1) * C + c] = (half_t)v;
}

// layer3 apply: IN+lrelu -> f16 NHWC with 65x65 zero pad (top-left origin) for convT4 gather
__global__ void inorm_lrelu_x3pad_k(const float* __restrict__ y3, const float* __restrict__ mean,
                                    const float* __restrict__ rstd, half_t* __restrict__ x3p) {
    int idx = blockIdx.x * 256 + threadIdx.x;            // 8*128*64*64 = 2^22
    if (idx >= 8 * 128 * 64 * 64) return;
    int xw = idx & 63, yh = (idx >> 6) & 63, c = (idx >> 12) & 127, b = idx >> 19;
    int bc = b * 128 + c;
    float v = (y3[idx] - mean[bc]) * rstd[bc];
    v = v > 0.f ? v : 0.2f * v;
    x3p[(((long)(b * 65 + yh) * 65) + xw) * 128 + c] = (half_t)v;
}

// layer4 apply: IN+lrelu -> f16 NHWC for conv5 WMMA
__global__ void inorm_lrelu_nhwc_h_k(const float* __restrict__ y4, const float* __restrict__ mean,
                                     const float* __restrict__ rstd, half_t* __restrict__ xh) {
    int idx = blockIdx.x * 256 + threadIdx.x;
    if (idx >= 8 * 256 * 128 * 128) return;
    int xw = idx & 127, yh = (idx >> 7) & 127, c = (idx >> 14) & 255, b = idx >> 22;
    int bc = b * 256 + c;
    float v = (y4[idx] - mean[bc]) * rstd[bc];
    v = v > 0.f ? v : 0.2f * v;
    xh[(((long)(b * 128 + yh) * 128) + xw) * 256 + c] = (half_t)v;
}

// ---------------- weight repacks ----------------
// w5 [512,256,3,3] -> f16 [N=512][K=2304], K = tap*256 + c
__global__ void repack_w5_k(const float* __restrict__ w5, half_t* __restrict__ wBt) {
    int idx = blockIdx.x * 256 + threadIdx.x;
    if (idx >= 512 * 2304) return;
    int k = idx % 2304, n = idx / 2304;
    int tap = k >> 8, c = k & 255, ky = tap / 3, kx = tap % 3;
    wBt[idx] = (half_t)w5[((n * 256 + c) * 3 + ky) * 3 + kx];
}

// generic OIHW k3 -> f16 [N=Cout][K=9*Cin], k = tap*Cin + ci  (for conv2/conv3)
__global__ void repack_oihw_k(const float* __restrict__ w, half_t* __restrict__ wB,
                              int Cin, int Cout) {
    int K = 9 * Cin;
    int total = Cout * K;
    int idx = blockIdx.x * 256 + threadIdx.x;
    if (idx >= total) return;
    int k = idx % K, n = idx / K;
    int tap = k / Cin, ci = k % Cin;
    int ky = tap / 3, kx = tap % 3;
    wB[idx] = (half_t)w[((n * Cin + ci) * 3 + ky) * 3 + kx];
}

__constant__ int c_kyT[4][4] = {{1,0,0,0},{1,1,0,0},{0,2,0,0},{0,0,2,2}};
__constant__ int c_kxT[4][4] = {{1,0,0,0},{0,2,0,0},{1,1,0,0},{0,2,0,2}};
__constant__ int c_ntT[4]    = {1,2,2,4};
__constant__ int c_cbT[4]    = {0,32768,98304,163840}; // class base offsets (halves)

// w4 [Cin=128][Cout=256][3,3] -> per-parity-class f16 [N=256][K=128*ntaps], k = tap*128 + ci
__global__ void repack_w4_k(const float* __restrict__ w4, half_t* __restrict__ wTt) {
    int idx = blockIdx.x * 256 + threadIdx.x;
    if (idx >= 294912) return;
    int cls, local;
    if      (idx <  32768) { cls = 0; local = idx; }
    else if (idx <  98304) { cls = 1; local = idx - 32768; }
    else if (idx < 163840) { cls = 2; local = idx - 98304; }
    else                   { cls = 3; local = idx - 163840; }
    int K = 128 * c_ntT[cls];
    int n = local / K, k = local % K;
    int tapi = k >> 7, ci = k & 127;
    int ky = c_kyT[cls][tapi], kx = c_kxT[cls][tapi];
    wTt[idx] = (half_t)w4[((ci * 256 + n) * 3 + ky) * 3 + kx];
}

// ---------------- shared WMMA tiling parameters ----------------
#define LDSA_S 40
#define LDSB_S 40

// ---------------- stride-2 conv as implicit GEMM via WMMA (conv2, conv3) ----------------
// M tile = 64 output pixels along x, N tile = COUT. 8 waves as 2(M) x 4(N).
// K = 9*CIN; each 32-wide K-step is one contiguous channel chunk of one tap.
template <int CIN, int COUT, int HOUT>
__global__ void __launch_bounds__(256) conv_s2_wmma_k(const half_t* __restrict__ xp,
                                                      const half_t* __restrict__ wB,
                                                      const float* __restrict__ bias,
                                                      float* __restrict__ y) {
    constexpr int HP    = 2 * HOUT + 2;     // padded input dim
    constexpr int CSTEP = CIN / 32;         // K substeps per tap
    constexpr int NSTEP = 9 * CSTEP;
    constexpr int NB    = COUT / 64;        // B frags per wave
    constexpr int TOPS  = 1 + ((COUT == 64) ? 1 : 2);  // async ops / thread / step
    constexpr int XT    = HOUT / 64;
    __shared__ half_t lA[2][64 * LDSA_S];
    __shared__ half_t lB[2][COUT * LDSB_S];
    int bx = blockIdx.x;
    int xt = bx % XT; int t = bx / XT;
    int oy = t % HOUT; int b = t / HOUT;
    int x0 = xt * 64;
    int tid = threadIdx.x, lane = tid & 31, wave = tid >> 5;
    int m0w = (wave & 1) * 32, n0w = (wave >> 1) * (COUT / 4);
    bool hi = lane >= 16; int l16 = lane & 15;
    v8f acc[2][NB] = {};
    int u = tid >> 2, v = tid & 3;

    auto issue = [&](int buf, int step) {
        int tap = step / CSTEP, cg = (step % CSTEP) * 32;
        int ky = tap / 3, kx = tap % 3;
        int iyp = 2 * oy + ky, ixp = 2 * (x0 + u) + kx;     // padded coords, always valid
        const half_t* asrc = xp + (((long)(b * HP + iyp) * HP + ixp) * CIN + cg + v * 8);
        CP16(&lA[buf][u * LDSA_S + v * 8], asrc);
        if constexpr (COUT == 64) {
            int n = tid >> 2, o = (tid & 3) * 8;
            CP16(&lB[buf][n * LDSB_S + o], wB + (long)n * (9 * CIN) + step * 32 + o);
        } else {
            int n = tid >> 1, o = (tid & 1) * 16;
            const half_t* bs = wB + (long)n * (9 * CIN) + step * 32 + o;
            CP16(&lB[buf][n * LDSB_S + o],     bs);
            CP16(&lB[buf][n * LDSB_S + o + 8], bs + 8);
        }
    };
    auto compute = [&](int buf) {
        V16U ua[2];
#pragma unroll
        for (int ma = 0; ma < 2; ++ma) {
            const half_t* ab = &lA[buf][(m0w + ma * 16 + l16) * LDSA_S + (hi ? 8 : 0)];
            ua[ma].h[0] = *(const v8h*)ab;
            ua[ma].h[1] = *(const v8h*)(ab + 16);
        }
#pragma unroll
        for (int jb = 0; jb < NB; ++jb) {
            const half_t* bb = &lB[buf][(n0w + jb * 16 + l16) * LDSB_S + (hi ? 8 : 0)];
            V16U ub; ub.h[0] = *(const v8h*)bb; ub.h[1] = *(const v8h*)(bb + 16);
#pragma unroll
            for (int ma = 0; ma < 2; ++ma)
                acc[ma][jb] = __builtin_amdgcn_wmma_f32_16x16x32_f16(
                    false, ua[ma].v, false, ub.v, (short)0, acc[ma][jb], false, false);
        }
    };

    issue(0, 0);
    int step = 0;
    for (; step < NSTEP - 1; ++step) {
        int cur = step & 1;
        issue(cur ^ 1, step + 1);
        ASYNC_WAIT_KEEP(TOPS);
        __syncthreads();
        compute(cur);
        __syncthreads();
    }
    ASYNC_WAIT_ALL();
    __syncthreads();
    compute(step & 1);

#pragma unroll
    for (int ma = 0; ma < 2; ++ma)
#pragma unroll
        for (int jb = 0; jb < NB; ++jb) {
            int co = n0w + jb * 16 + l16;
            float bv = bias[co];
            int xbase = x0 + m0w + ma * 16 + (hi ? 8 : 0);
            float* dst = y + (((long)(b * COUT + co) * HOUT + oy) * HOUT + xbase);
#pragma unroll
            for (int r = 0; r < 8; ++r) dst[r] = acc[ma][jb][r] + bv;
        }
}

// ---------------- convT4 as 4 parity-class GEMMs via WMMA ----------------
__global__ void __launch_bounds__(256) convt4_wmma_k(const half_t* __restrict__ x3p,
                                                     const half_t* __restrict__ wTt,
                                                     const float* __restrict__ b4,
                                                     float* __restrict__ y4) {
    __shared__ half_t lA[2][64 * LDSA_S];
    __shared__ half_t lB[2][256 * LDSB_S];
    int bx = blockIdx.x;                        // b*256 + oy*2 + px
    int px = bx & 1, oy = (bx >> 1) & 127, b = bx >> 8;
    int py = oy & 1;
    int cls = py * 2 + px;
    int nt = c_ntT[cls];
    int nstep = nt * 4;                         // K = 128*nt, steps of 32
    long cbase = c_cbT[cls];
    int rowK = 128 * nt;

    int tid = threadIdx.x, lane = tid & 31, wave = tid >> 5;
    int m0w = (wave & 1) * 32, n0w = (wave >> 1) * 64;
    bool hi = lane >= 16; int l16 = lane & 15;
    v8f acc[2][4] = {};
    int u = tid >> 2, v = tid & 3;              // A stage: pixel u, channel octet v

    auto issue = [&](int buf, int step) {
        int tapi = step >> 2, cg = (step & 3) * 32;
        int ky = c_kyT[cls][tapi], kx = c_kxT[cls][tapi];
        int iy = (oy + 1 - ky) >> 1;            // parity-exact by construction
        int ix = ((2 * u + px) + 1 - kx) >> 1;  // u or u+1 (pad row/col holds zeros)
        const half_t* asrc = x3p + (((long)(b * 65 + iy) * 65 + ix) * 128 + cg + v * 8);
        CP16(&lA[buf][u * LDSA_S + v * 8], asrc);
        const half_t* bsrc = wTt + cbase + (long)tid * rowK + step * 32;
        half_t* bdst = &lB[buf][tid * LDSB_S];
        CP16(bdst,      bsrc);
        CP16(bdst + 8,  bsrc + 8);
        CP16(bdst + 16, bsrc + 16);
        CP16(bdst + 24, bsrc + 24);
    };
    auto compute = [&](int buf) {
        V16U ua[2];
#pragma unroll
        for (int ma = 0; ma < 2; ++ma) {
            const half_t* ab = &lA[buf][(m0w + ma * 16 + l16) * LDSA_S + (hi ? 8 : 0)];
            ua[ma].h[0] = *(const v8h*)ab;
            ua[ma].h[1] = *(const v8h*)(ab + 16);
        }
#pragma unroll
        for (int j = 0; j < 4; ++j) {
            const half_t* bb = &lB[buf][(n0w + j * 16 + l16) * LDSB_S + (hi ? 8 : 0)];
            V16U ub; ub.h[0] = *(const v8h*)bb; ub.h[1] = *(const v8h*)(bb + 16);
#pragma unroll
            for (int ma = 0; ma < 2; ++ma)
                acc[ma][j] = __builtin_amdgcn_wmma_f32_16x16x32_f16(
                    false, ua[ma].v, false, ub.v, (short)0, acc[ma][j], false, false);
        }
    };

    issue(0, 0);
    int step = 0;
    for (; step < nstep - 1; ++step) {
        int cur = step & 1;
        issue(cur ^ 1, step + 1);
        ASYNC_WAIT_KEEP(5);
        __syncthreads();
        compute(cur);
        __syncthreads();
    }
    ASYNC_WAIT_ALL();
    __syncthreads();
    compute(step & 1);

    // D layout: VGPR r, lanes 0-15 -> (M=r, N=lane); lanes 16-31 -> (M=8+r, N=lane-16)
#pragma unroll
    for (int ma = 0; ma < 2; ++ma)
#pragma unroll
        for (int j = 0; j < 4; ++j) {
            int co = n0w + j * 16 + l16;
            float bias = b4[co];
            int ub = m0w + ma * 16 + (hi ? 8 : 0);
            float* dst = y4 + (((long)(b * 256 + co) * 128 + oy) * 128);
#pragma unroll
            for (int r = 0; r < 8; ++r) dst[2 * (ub + r) + px] = acc[ma][j][r] + bias;
        }
}

// ---------------- conv5 implicit GEMM via WMMA (K = 2304) ----------------
__global__ void __launch_bounds__(256) conv5_wmma_k(const half_t* __restrict__ xh,
                                                    const half_t* __restrict__ wBt,
                                                    const float* __restrict__ b5,
                                                    float* __restrict__ codes) {
    __shared__ half_t lA[2][64 * LDSA_S];
    __shared__ half_t lB[2][256 * LDSB_S];
    int bx = blockIdx.x;
    int xt = bx & 1, yh = (bx >> 1) & 127, b = bx >> 8;
    int n0 = blockIdx.y * 256;
    int x0 = xt * 64;
    int tid = threadIdx.x, lane = tid & 31, wave = tid >> 5;
    int m0w = (wave & 1) * 32, n0w = (wave >> 1) * 64;
    bool hi = lane >= 16; int l16 = lane & 15;
    v8f acc[2][4] = {};
    int u = tid >> 2, v = tid & 3;

    auto issue = [&](int buf, int step) {
        int tap = step >> 3, cg = (step & 7) * 32;
        int ty = tap / 3, tx = tap % 3;
        int ry = reflect_idx(yh + ty - 1, 128);
        int rx = reflect_idx(x0 + u + tx - 1, 128);
        const half_t* asrc = xh + (((long)(b * 128 + ry) * 128 + rx) * 256 + cg + v * 8);
        CP16(&lA[buf][u * LDSA_S + v * 8], asrc);
        const half_t* bsrc = wBt + ((long)(n0 + tid) * 2304 + step * 32);
        half_t* bdst = &lB[buf][tid * LDSB_S];
        CP16(bdst,      bsrc);
        CP16(bdst + 8,  bsrc + 8);
        CP16(bdst + 16, bsrc + 16);
        CP16(bdst + 24, bsrc + 24);
    };
    auto compute = [&](int buf) {
        V16U ua[2];
#pragma unroll
        for (int ma = 0; ma < 2; ++ma) {
            const half_t* ab = &lA[buf][(m0w + ma * 16 + l16) * LDSA_S + (hi ? 8 : 0)];
            ua[ma].h[0] = *(const v8h*)ab;
            ua[ma].h[1] = *(const v8h*)(ab + 16);
        }
#pragma unroll
        for (int j = 0; j < 4; ++j) {
            const half_t* bb = &lB[buf][(n0w + j * 16 + l16) * LDSB_S + (hi ? 8 : 0)];
            V16U ub; ub.h[0] = *(const v8h*)bb; ub.h[1] = *(const v8h*)(bb + 16);
#pragma unroll
            for (int ma = 0; ma < 2; ++ma)
                acc[ma][j] = __builtin_amdgcn_wmma_f32_16x16x32_f16(
                    false, ua[ma].v, false, ub.v, (short)0, acc[ma][j], false, false);
        }
    };

    issue(0, 0);
    int step = 0;
    for (; step < 71; ++step) {
        int cur = step & 1;
        issue(cur ^ 1, step + 1);
        ASYNC_WAIT_KEEP(5);
        __syncthreads();
        compute(cur);
        __syncthreads();
    }
    ASYNC_WAIT_ALL();
    __syncthreads();
    compute(step & 1);

#pragma unroll
    for (int ma = 0; ma < 2; ++ma)
#pragma unroll
        for (int j = 0; j < 4; ++j) {
            int ch = n0 + n0w + j * 16 + l16;
            float bias = b5[ch];
            int xbase = x0 + m0w + ma * 16 + (hi ? 8 : 0);
            float* dst = codes + (((long)(b * 512 + ch) * 128 + yh) * 128 + xbase);
#pragma unroll
            for (int r = 0; r < 8; ++r) dst[r] = fast_tanh(acc[ma][j][r] + bias);
        }
}

// ---------------- pooling over segmentation classes ----------------
__global__ void labels_k(const float* __restrict__ seg, int* __restrict__ labels) {
    int idx = blockIdx.x * 256 + threadIdx.x;            // 8*128*128
    if (idx >= 8 * 128 * 128) return;
    int xw = idx & 127, yh = (idx >> 7) & 127, b = idx >> 14;
    int lab = 0;
    for (int s = 0; s < 19; ++s)
        if (seg[(((long)(b * 19 + s) * 256) + 2 * yh) * 256 + 2 * xw] > 0.5f) lab = s;
    labels[idx] = lab;
}

__global__ void zero_k(float* __restrict__ p, long n) {
    long i = (long)blockIdx.x * 256 + threadIdx.x;
    if (i < n) p[i] = 0.f;
}

__global__ void __launch_bounds__(256) region_sum_k(const float* __restrict__ codes,
                                                    const int* __restrict__ labels,
                                                    float* __restrict__ sums,
                                                    float* __restrict__ counts) {
    __shared__ int lsl[16 * 128];
    int b = blockIdx.x, y0 = blockIdx.y * 16;
    int tid = threadIdx.x;
    for (int i = tid; i < 2048; i += 256)
        lsl[i] = labels[(b * 128 + y0 + (i >> 7)) * 128 + (i & 127)];
    __syncthreads();
    int c0 = tid * 2;
    float acc0[19], acc1[19];
    for (int s = 0; s < 19; ++s) { acc0[s] = 0.f; acc1[s] = 0.f; }
    const float* p0 = codes + (long)(b * 512 + c0) * 128 * 128;
    const float* p1 = p0 + 128 * 128;
    for (int yy = 0; yy < 16; ++yy) {
        long rowoff = (long)(y0 + yy) * 128;
        for (int xx = 0; xx < 128; ++xx) {
            int lab = lsl[yy * 128 + xx];
            acc0[lab] += p0[rowoff + xx];
            acc1[lab] += p1[rowoff + xx];
        }
    }
    for (int s = 0; s < 19; ++s) {
        atomicAdd(&sums[(b * 19 + s) * 512 + c0], acc0[s]);
        atomicAdd(&sums[(b * 19 + s) * 512 + c0 + 1], acc1[s]);
    }
    if (tid < 19) {
        int cnt = 0;
        for (int i = 0; i < 2048; ++i) cnt += (lsl[i] == tid);
        atomicAdd(&counts[b * 19 + tid], (float)cnt);
    }
}

__global__ void finalize_k(const float* __restrict__ sums, const float* __restrict__ counts,
                           float* __restrict__ out) {
    int idx = blockIdx.x * 256 + threadIdx.x;
    if (idx >= 8 * 19 * 512) return;
    int bs = idx >> 9;
    float c = counts[bs];
    out[idx] = c > 0.f ? sums[idx] / fmaxf(c, 1.f) : 0.f;
}

// ---------------- host-side launcher ----------------
extern "C" void kernel_launch(void* const* d_in, const int* in_sizes, int n_in,
                              void* d_out, int out_size, void* d_ws, size_t ws_size,
                              hipStream_t stream) {
    const float* input  = (const float*)d_in[0];
    const float* segmap = (const float*)d_in[1];
    const float* w1 = (const float*)d_in[2];  const float* b1 = (const float*)d_in[3];
    const float* w2 = (const float*)d_in[4];  const float* b2 = (const float*)d_in[5];
    const float* w3 = (const float*)d_in[6];  const float* b3 = (const float*)d_in[7];
    const float* w4 = (const float*)d_in[8];  const float* b4 = (const float*)d_in[9];
    const float* w5 = (const float*)d_in[10]; const float* b5 = (const float*)d_in[11];
    float* out = (float*)d_out;

    char* ws = (char*)d_ws;
    size_t off = 0;
    auto take = [&](size_t bytes) { char* p = ws + off; off += (bytes + 255) & ~(size_t)255; return p; };
    float*  y1     = (float*)take((size_t)8 * 32 * 256 * 256 * 4);   // 64 MB
    half_t* x1p    = (half_t*)take((size_t)8 * 258 * 258 * 32 * 2);  // 34 MB (zero-padded NHWC)
    float*  y2     = (float*)take((size_t)8 * 64 * 128 * 128 * 4);   // 32 MB
    half_t* x2p    = (half_t*)take((size_t)8 * 130 * 130 * 64 * 2);  // 17 MB (zero-padded NHWC)
    float*  y3     = (float*)take((size_t)8 * 128 * 64 * 64 * 4);    // 16 MB
    half_t* x3p    = (half_t*)take((size_t)8 * 65 * 65 * 128 * 2);   // 8.7 MB (zero-padded)
    float*  y4     = (float*)take((size_t)8 * 256 * 128 * 128 * 4);  // 128 MB
    half_t* x4h    = (half_t*)take((size_t)8 * 256 * 128 * 128 * 2); // 64 MB
    half_t* wB2    = (half_t*)take((size_t)64 * 288 * 2);
    half_t* wB3    = (half_t*)take((size_t)128 * 576 * 2);
    half_t* wBt    = (half_t*)take((size_t)512 * 2304 * 2);          // 2.3 MB
    half_t* wTt    = (half_t*)take((size_t)294912 * 2);              // 0.6 MB
    int*    labels = (int*)take((size_t)8 * 128 * 128 * 4);
    float*  meanb  = (float*)take((size_t)2048 * 4);
    float*  rstdb  = (float*)take((size_t)2048 * 4);
    float*  sums   = (float*)take((size_t)8 * 19 * 512 * 4);
    float*  counts = (float*)take((size_t)256 * 4);
    float*  codes  = (float*)take((size_t)8 * 512 * 128 * 128 * 4);  // 256 MB

    // zero accumulators and padded activation buffers
    zero_k<<<(8 * 19 * 512 + 255) / 256, 256, 0, stream>>>(sums, 8 * 19 * 512);
    zero_k<<<1, 256, 0, stream>>>(counts, 256);
    long z1 = (long)8 * 258 * 258 * 32 * 2 / 4;
    zero_k<<<(int)((z1 + 255) / 256), 256, 0, stream>>>((float*)x1p, z1);
    long z2 = (long)8 * 130 * 130 * 64 * 2 / 4;
    zero_k<<<(int)((z2 + 255) / 256), 256, 0, stream>>>((float*)x2p, z2);
    long z3 = (long)8 * 65 * 65 * 128 * 2 / 4;
    zero_k<<<(int)((z3 + 255) / 256), 256, 0, stream>>>((float*)x3p, z3);

    repack_oihw_k<<<(64 * 288 + 255) / 256, 256, 0, stream>>>(w2, wB2, 32, 64);
    repack_oihw_k<<<(128 * 576 + 255) / 256, 256, 0, stream>>>(w3, wB3, 64, 128);
    repack_w4_k<<<(294912 + 255) / 256, 256, 0, stream>>>(w4, wTt);
    repack_w5_k<<<(512 * 2304 + 255) / 256, 256, 0, stream>>>(w5, wBt);

    // conv1 + IN + lrelu -> padded f16 NHWC
    conv1_k<<<(8 * 32 * 256 * 256) / 256, 256, 0, stream>>>(input, w1, b1, y1);
    inorm_stats_k<<<8 * 32, 256, 0, stream>>>(y1, meanb, rstdb, 256 * 256);
    inorm_lrelu_pad_k<<<(8 * 32 * 256 * 256) / 256, 256, 0, stream>>>(
        y1, meanb, rstdb, x1p, 32, 256, (long)8 * 32 * 256 * 256);

    // conv2 s2 (WMMA) + IN + lrelu -> padded f16 NHWC
    conv_s2_wmma_k<32, 64, 128><<<8 * 128 * 2, 256, 0, stream>>>(x1p, wB2, b2, y2);
    inorm_stats_k<<<8 * 64, 256, 0, stream>>>(y2, meanb, rstdb, 128 * 128);
    inorm_lrelu_pad_k<<<(8 * 64 * 128 * 128) / 256, 256, 0, stream>>>(
        y2, meanb, rstdb, x2p, 64, 128, (long)8 * 64 * 128 * 128);

    // conv3 s2 (WMMA) + IN + lrelu -> padded f16 NHWC (65x65 layout for convT4)
    conv_s2_wmma_k<64, 128, 64><<<8 * 64, 256, 0, stream>>>(x2p, wB3, b3, y3);
    inorm_stats_k<<<8 * 128, 256, 0, stream>>>(y3, meanb, rstdb, 64 * 64);
    inorm_lrelu_x3pad_k<<<(8 * 128 * 64 * 64) / 256, 256, 0, stream>>>(y3, meanb, rstdb, x3p);

    // convT4 (WMMA, parity decomposition) + IN + lrelu -> f16 NHWC
    convt4_wmma_k<<<8 * 128 * 2, 256, 0, stream>>>(x3p, wTt, b4, y4);
    inorm_stats_k<<<8 * 256, 256, 0, stream>>>(y4, meanb, rstdb, 128 * 128);
    inorm_lrelu_nhwc_h_k<<<(8 * 256 * 128 * 128) / 256, 256, 0, stream>>>(y4, meanb, rstdb, x4h);

    // conv5 (WMMA implicit GEMM) + bias + tanh
    dim3 gridC(8 * 128 * 2, 2);
    conv5_wmma_k<<<gridC, 256, 0, stream>>>(x4h, wBt, b5, codes);

    // pooling
    labels_k<<<(8 * 128 * 128) / 256, 256, 0, stream>>>(segmap, labels);
    dim3 gridS(8, 8);
    region_sum_k<<<gridS, 256, 0, stream>>>(codes, labels, sums, counts);
    finalize_k<<<(8 * 19 * 512 + 255) / 256, 256, 0, stream>>>(sums, counts, out);
}
